// VisionMamba_19104014532636
// MI455X (gfx1250) — matrix operand order repaired and checked by reading, not compiled, not run on hardware
//
#include <hip/hip_runtime.h>
#include <hip/hip_bf16.h>
#include <math.h>

// ---------------------------------------------------------------------------
// Vision Mamba forward for MI455X (gfx1250), bf16 WMMA GEMMs + scalar scan.
// GEMM tiles are staged into LDS directly in the WMMA fragment layout so the
// inner loop is pure b128 traffic: global_load_b128 -> v_cvt_pk_bf16_f32 ->
// ds_store_b128 -> ds_load_b128 -> v_wmma_f32_16x16x32_bf16 (x4 per wave).
// ---------------------------------------------------------------------------

typedef __attribute__((ext_vector_type(16))) __bf16 v16bf;
typedef __attribute__((ext_vector_type(8)))  float  v8f;
typedef __attribute__((ext_vector_type(2)))  float  v2f_t;
typedef __attribute__((ext_vector_type(2)))  __bf16 v2bf_t;

static constexpr int kDim    = 768;
static constexpr int kDI     = 1536;
static constexpr int kDtRank = 48;
static constexpr int kNState = 16;
static constexpr int kDepth  = 12;
static constexpr int kL      = 197;   // 196 patches + cls
static constexpr int kB      = 4;
static constexpr int kRows   = kB * kL;               // 788
static constexpr int kXDbl   = kDtRank + 2 * kNState; // 80

enum { EP_NONE = 0, EP_BIAS = 1, EP_BIAS_SOFTPLUS = 2, EP_BIAS_RESID = 3 };

union FragU { uint4 q[2]; v16bf v; };   // 32 bytes: one 8-VGPR bf16 fragment

// pack two fp32 -> one dword of two bf16 (v_cvt_pk_bf16_f32, RNE)
static __device__ __forceinline__ unsigned pk2(float lo, float hi)
{
    v2f_t t; t.x = lo; t.y = hi;
    v2bf_t r = __builtin_convertvector(t, v2bf_t);
    union { v2bf_t b; unsigned u; } c; c.b = r;
    return c.u;
}

// ---------------------------------------------------------------------------
// GEMM: C[M,N] = A[M,K] * W[N,K]^T  (W row-major (N,K), einsum 'bld,ed->ble').
// 256 threads = 8 waves; 64x128 C tile per block; K stepped by 32.
// Each wave owns one 16-row M tile and four 16-col N tiles (4 accumulators,
// 4 v_wmma per K step).  Epilogues: +bias[col], softplus, +residual; optional
// per-Lr row reversal and column offset (bwd direction / concat halves).
// ---------------------------------------------------------------------------
__global__ __launch_bounds__(256)
void gemm_wmma_bf16(const float* __restrict__ A, int lda,
                    const float* __restrict__ W,
                    float* __restrict__ C, int ldc, int col_off,
                    const float* __restrict__ bias,
                    const float* __restrict__ resid, int ldr,
                    int M, int N, int K,
                    int ep, int rev, int Lr)
{
    // Fragment-layout LDS: [tile][lane 0..31][2 x uint4]
    __shared__ uint4 AsQ[4 * 32 * 2];   // 4 M tiles  (4 KB)
    __shared__ uint4 BsQ[8 * 32 * 2];   // 8 N tiles  (8 KB)

    const int tid   = threadIdx.x;
    const int wave  = tid >> 5;
    const int lane  = tid & 31;
    const int lm    = lane & 15;
    const int lg    = lane >> 4;
    const int m0    = blockIdx.y * 64;
    const int n0    = blockIdx.x * 128;
    const int mt    = wave & 3;    // wave's M tile (0..3)
    const int npair = wave >> 2;   // 0/1 -> n tiles {0..3} or {4..7}

    // Staging assignment: thread = (sm, sq); handles row sm of a 64-row slab,
    // fp32 elements k = 8*sq .. 8*sq+7 (4 bf16-pair dwords, one uint4).
    const int sm = tid >> 2;
    const int sq = tid & 3;
    // A-fragment inverse map: pair kp -> lanegrp g=(kp>>2)&1, vgpr i=(kp&3)+4*(kp>>3)
    //   4 consecutive pairs (4sq..4sq+3) => g=sq&1, uint4 slot = sq>>1.
    const int aslot = ((sm >> 4) * 32 + ((sq & 1) * 16 + (sm & 15))) * 2 + (sq >> 1);
    // B-fragment inverse map: same as A => g=sq&1, uint4 slot = sq>>1.
    const int bslot = ((sm >> 4) * 32 + ((sq & 1) * 16 + (sm & 15))) * 2 + (sq >> 1);

    const int arow  = m0 + sm;
    const int brow0 = n0 + sm;        // B slab 0 (n tiles 0..3)
    const int brow1 = n0 + sm + 64;   // B slab 1 (n tiles 4..7)

    v8f c[4];
#pragma unroll
    for (int t = 0; t < 4; ++t) c[t] = (v8f){0.f,0.f,0.f,0.f,0.f,0.f,0.f,0.f};

    for (int k0 = 0; k0 < K; k0 += 32) {
        const int kk = k0 + 8 * sq;
        const bool kok = (kk + 8 <= K);
        float4 a0 = {0.f,0.f,0.f,0.f}, a1 = a0;
        float4 e0 = a0, e1 = a0, f0 = a0, f1 = a0;
        if (arow < M && kok) {
            const float4* p = (const float4*)(A + (long)arow * lda + kk);
            a0 = p[0]; a1 = p[1];
        }
        if (brow0 < N && kok) {
            const float4* p = (const float4*)(W + (long)brow0 * K + kk);
            e0 = p[0]; e1 = p[1];
        }
        if (brow1 < N && kok) {
            const float4* p = (const float4*)(W + (long)brow1 * K + kk);
            f0 = p[0]; f1 = p[1];
        }
        if (k0 + 32 < K) {   // global_prefetch_b8 for the next K tile
            if (arow  < M) __builtin_prefetch(A + (long)arow  * lda + k0 + 32, 0, 0);
            if (brow0 < N) __builtin_prefetch(W + (long)brow0 * K   + k0 + 32, 0, 0);
            if (brow1 < N) __builtin_prefetch(W + (long)brow1 * K   + k0 + 32, 0, 0);
        }
        uint4 pa, pb0, pb1;
        pa.x  = pk2(a0.x, a0.y); pa.y  = pk2(a0.z, a0.w);
        pa.z  = pk2(a1.x, a1.y); pa.w  = pk2(a1.z, a1.w);
        pb0.x = pk2(e0.x, e0.y); pb0.y = pk2(e0.z, e0.w);
        pb0.z = pk2(e1.x, e1.y); pb0.w = pk2(e1.z, e1.w);
        pb1.x = pk2(f0.x, f0.y); pb1.y = pk2(f0.z, f0.w);
        pb1.z = pk2(f1.x, f1.y); pb1.w = pk2(f1.z, f1.w);

        __syncthreads();              // previous iteration's reads done
        AsQ[aslot]       = pa;        // ds_store_b128 each
        BsQ[bslot]       = pb0;
        BsQ[bslot + 256] = pb1;       // +4 tiles * 32 lanes * 2
        __syncthreads();

        FragU fa;
        const int ab = (mt * 32 + lane) * 2;
        fa.q[0] = AsQ[ab]; fa.q[1] = AsQ[ab + 1];
#pragma unroll
        for (int t = 0; t < 4; ++t) {
            FragU fb;
            const int bb = ((npair * 4 + t) * 32 + lane) * 2;
            fb.q[0] = BsQ[bb]; fb.q[1] = BsQ[bb + 1];
            c[t] = __builtin_amdgcn_wmma_f32_16x16x32_bf16(false, fa.v, false, fb.v,
                                                           (short)0, c[t], false, false);
        }
    }

    // Store: C/D layout -> VGPR i holds row (i + 8*lanegroup), col lane%16.
#pragma unroll
    for (int i = 0; i < 8; ++i) {
        int row = m0 + mt * 16 + lg * 8 + i;
        if (row >= M) continue;
        int orow = row;
        if (rev) { int seg = row / Lr, l = row - seg * Lr; orow = seg * Lr + (Lr - 1 - l); }
#pragma unroll
        for (int t = 0; t < 4; ++t) {
            int col = n0 + (npair * 4 + t) * 16 + lm;
            if (col >= N) continue;
            float v = c[t][i];
            if (ep != EP_NONE) v += bias[col];
            if (ep == EP_BIAS_SOFTPLUS) v = (v > 20.f) ? v : log1pf(__expf(v));
            if (ep == EP_BIAS_RESID) v += resid[(long)orow * ldr + col];
            C[(long)orow * ldc + col_off + col] = v;
        }
    }
}

// ---------------------------------------------------------------------------
// LayerNorm over 768 features: one block per row, 3 elems/thread.
// ---------------------------------------------------------------------------
__global__ __launch_bounds__(256)
void ln_kernel(const float* __restrict__ x, long xstride,
               const float* __restrict__ w, const float* __restrict__ b,
               float* __restrict__ y, long ystride)
{
    __shared__ float s1[256], s2[256];
    long r = blockIdx.x;
    const float* xr = x + r * xstride;
    float v[3];
    float lsum = 0.f, lsq = 0.f;
#pragma unroll
    for (int j = 0; j < 3; ++j) {
        v[j] = xr[threadIdx.x + 256 * j];
        lsum += v[j]; lsq += v[j] * v[j];
    }
    s1[threadIdx.x] = lsum; s2[threadIdx.x] = lsq;
    __syncthreads();
    for (int o = 128; o > 0; o >>= 1) {
        if ((int)threadIdx.x < o) { s1[threadIdx.x] += s1[threadIdx.x + o];
                                    s2[threadIdx.x] += s2[threadIdx.x + o]; }
        __syncthreads();
    }
    float mean = s1[0] * (1.f / 768.f);
    float var  = s2[0] * (1.f / 768.f) - mean * mean;
    float rstd = rsqrtf(var + 1e-5f);
#pragma unroll
    for (int j = 0; j < 3; ++j) {
        int c = threadIdx.x + 256 * j;
        y[r * ystride + c] = (v[j] - mean) * rstd * w[c] + b[c];
    }
}

// ---------------------------------------------------------------------------
// Patch rearrange: x(4,4,224,224) -> A(784, 1024) with cols ordered (C,Pi,Pj)
// ---------------------------------------------------------------------------
__global__ __launch_bounds__(256)
void rearrange_kernel(const float* __restrict__ x, float* __restrict__ pa)
{
    int idx = blockIdx.x * 256 + threadIdx.x;
    if (idx >= kB * 196 * 1024) return;
    int b = idx / (196 * 1024);
    int rem = idx % (196 * 1024);
    int p = rem >> 10, q = rem & 1023;
    int c = q >> 8, pi = (q >> 4) & 15, pj = q & 15;
    int ph = p / 14, pw = p % 14;
    pa[idx] = x[(((long)b * 4 + c) * 224 + ph * 16 + pi) * 224 + pw * 16 + pj];
}

// h[b,0,:] = cls + pos[0];  h[b,l,:] = pe_ln[b,l-1,:] + pos[l]
__global__ __launch_bounds__(256)
void assemble_kernel(const float* __restrict__ pe, const float* __restrict__ cls,
                     const float* __restrict__ pos, float* __restrict__ h)
{
    int idx = blockIdx.x * 256 + threadIdx.x;
    if (idx >= kRows * kDim) return;
    int r = idx / kDim, d = idx % kDim;
    int b = r / kL, l = r % kL;
    float v = (l == 0) ? cls[d] : pe[((long)b * 196 + (l - 1)) * kDim + d];
    h[idx] = v + pos[(long)l * kDim + d];
}

// Reverse sequence within each batch: out[b,l,:] = in[b,L-1-l,:]
__global__ __launch_bounds__(256)
void reverse_kernel(const float* __restrict__ in, float* __restrict__ out)
{
    int idx = blockIdx.x * 256 + threadIdx.x;
    if (idx >= kRows * kDim) return;
    int r = idx / kDim, d = idx % kDim;
    int b = r / kL, l = r % kL;
    out[idx] = in[((long)b * kL + (kL - 1 - l)) * kDim + d];
}

// Causal depthwise conv1d (width 4) over u half of xz, + bias, SiLU.
__global__ __launch_bounds__(256)
void conv_silu_kernel(const float* __restrict__ xz,
                      const float* __restrict__ cw,  // (DI,4)
                      const float* __restrict__ cb,  // (DI)
                      float* __restrict__ u)
{
    int idx = blockIdx.x * 256 + threadIdx.x;
    if (idx >= kRows * kDI) return;
    int r = idx / kDI, e = idx % kDI;
    int b = r / kL, l = r % kL;
    float acc = cb[e];
#pragma unroll
    for (int k = 0; k < 4; ++k) {
        int lp = l + k - 3;
        if (lp >= 0)
            acc += xz[((long)b * kL + lp) * (2 * kDI) + e] * cw[e * 4 + k];
    }
    u[idx] = acc / (1.f + __expf(-acc));   // SiLU
}

// Selective scan: one thread per (batch, channel). Sequential over L.
__global__ __launch_bounds__(256)
void scan_kernel(const float* __restrict__ u, const float* __restrict__ dt,
                 const float* __restrict__ xdbl,   // (B,L,80): [48:64]=B, [64:80]=C
                 const float* __restrict__ A_log,  // (DI,16)
                 float* __restrict__ y)
{
    int e = blockIdx.x * 256 + threadIdx.x;   // channel
    int b = blockIdx.y;
    if (e >= kDI) return;
    float Ar[kNState], h[kNState];
#pragma unroll
    for (int n = 0; n < kNState; ++n) {
        Ar[n] = -__expf(A_log[(long)e * kNState + n]);
        h[n] = 0.f;
    }
    for (int l = 0; l < kL; ++l) {
        long row = (long)b * kL + l;
        const float4* bp = (const float4*)(xdbl + row * kXDbl + kDtRank);
        float Bt[kNState], Ct[kNState];
        *(float4*)(Bt + 0)  = bp[0]; *(float4*)(Bt + 4)  = bp[1];
        *(float4*)(Bt + 8)  = bp[2]; *(float4*)(Bt + 12) = bp[3];
        *(float4*)(Ct + 0)  = bp[4]; *(float4*)(Ct + 4)  = bp[5];
        *(float4*)(Ct + 8)  = bp[6]; *(float4*)(Ct + 12) = bp[7];
        float ut  = u[row * kDI + e];
        float dtt = dt[row * kDI + e];
        float du  = dtt * ut;
        float acc = 0.f;
#pragma unroll
        for (int n = 0; n < kNState; ++n) {
            float dA = __expf(dtt * Ar[n]);
            h[n] = h[n] * dA + du * Bt[n];
            acc += h[n] * Ct[n];
        }
        y[row * kDI + e] = acc;
    }
}

// y = (y_scan + u*Dp) * silu(z),  z = xz[..., DI + e]
__global__ __launch_bounds__(256)
void gate_kernel(float* __restrict__ y, const float* __restrict__ u,
                 const float* __restrict__ xz, const float* __restrict__ Dp)
{
    int idx = blockIdx.x * 256 + threadIdx.x;
    if (idx >= kRows * kDI) return;
    int r = idx / kDI, e = idx % kDI;
    float z = xz[(long)r * (2 * kDI) + kDI + e];
    float s = z / (1.f + __expf(-z));
    y[idx] = (y[idx] + u[idx] * Dp[e]) * s;
}

// ---------------------------------------------------------------------------
extern "C" void kernel_launch(void* const* d_in, const int* in_sizes, int n_in,
                              void* d_out, int out_size, void* d_ws, size_t ws_size,
                              hipStream_t stream)
{
    const float* x       = (const float*)d_in[0];
    const float* patch_W = (const float*)d_in[1];
    const float* patch_b = (const float*)d_in[2];
    const float* pe_ln_w = (const float*)d_in[3];
    const float* pe_ln_b = (const float*)d_in[4];
    const float* cls     = (const float*)d_in[5];
    const float* pos     = (const float*)d_in[6];
    const float* ln_w    = (const float*)d_in[7];
    const float* ln_b    = (const float*)d_in[8];
    const float* in_W    = (const float*)d_in[9];
    const float* conv_W  = (const float*)d_in[10];
    const float* conv_b  = (const float*)d_in[11];
    const float* xproj_W = (const float*)d_in[12];
    const float* dt_W    = (const float*)d_in[13];
    const float* dt_b    = (const float*)d_in[14];
    const float* A_log   = (const float*)d_in[15];
    const float* Dp      = (const float*)d_in[16];
    const float* out_W   = (const float*)d_in[17];
    const float* merge_W = (const float*)d_in[18];
    const float* merge_b = (const float*)d_in[19];
    const float* fn_w    = (const float*)d_in[20];
    const float* fn_b    = (const float*)d_in[21];
    float* out = (float*)d_out;

    // workspace carve-out (~43 MB of fp32)
    float* ws = (float*)d_ws;
    size_t off = 0;
    auto alloc = [&](size_t n) { float* p = ws + off; off += (n + 63) & ~(size_t)63; return p; };
    float* h_   = alloc((size_t)kRows * kDim);
    float* hn   = alloc((size_t)kRows * kDim);
    float* hnr  = alloc((size_t)kRows * kDim);
    float* pe_a = alloc((size_t)784 * 1024);
    float* pe_o = alloc((size_t)784 * kDim);
    float* xz   = alloc((size_t)kRows * 2 * kDI);
    float* uact = alloc((size_t)kRows * kDI);
    float* xdbl = alloc((size_t)kRows * kXDbl);
    float* dtb  = alloc((size_t)kRows * kDI);
    float* ysc  = alloc((size_t)kRows * kDI);
    float* cat  = alloc((size_t)kRows * kDI);

    dim3 blk(256);
    auto gemm = [&](const float* A, int lda, const float* W, float* C, int ldc,
                    int coff, const float* bias, const float* res, int ldr,
                    int M, int N, int K, int ep, int rev, int Lr) {
        dim3 grid((N + 127) / 128, (M + 63) / 64);
        gemm_wmma_bf16<<<grid, blk, 0, stream>>>(A, lda, W, C, ldc, coff,
                                                 bias, res, ldr, M, N, K, ep, rev, Lr);
    };

    // ---- patch embed ----
    rearrange_kernel<<<(kB * 196 * 1024 + 255) / 256, blk, 0, stream>>>(x, pe_a);
    gemm(pe_a, 1024, patch_W, pe_o, kDim, 0, patch_b, nullptr, 0,
         784, kDim, 1024, EP_BIAS, 0, 1);
    ln_kernel<<<784, blk, 0, stream>>>(pe_o, kDim, pe_ln_w, pe_ln_b, hn, kDim);
    assemble_kernel<<<(kRows * kDim + 255) / 256, blk, 0, stream>>>(hn, cls, pos, h_);

    // ---- 12 bidirectional Mamba layers ----
    for (int d = 0; d < kDepth; ++d) {
        ln_kernel<<<kRows, blk, 0, stream>>>(h_, kDim, ln_w + (long)d * kDim,
                                             ln_b + (long)d * kDim, hn, kDim);
        reverse_kernel<<<(kRows * kDim + 255) / 256, blk, 0, stream>>>(hn, hnr);

        for (int dir = 0; dir < 2; ++dir) {
            const float* X = dir ? hnr : hn;
            long wd = (long)d * 2 + dir;
            // in-proj: (788,768) x (3072,768)^T -> xz
            gemm(X, kDim, in_W + wd * 2 * kDI * kDim, xz, 2 * kDI, 0,
                 nullptr, nullptr, 0, kRows, 2 * kDI, kDim, EP_NONE, 0, 1);
            // causal dwconv + SiLU
            conv_silu_kernel<<<(kRows * kDI + 255) / 256, blk, 0, stream>>>(
                xz, conv_W + wd * kDI * 4, conv_b + wd * kDI, uact);
            // x-proj: (788,1536) x (80,1536)^T -> xdbl
            gemm(uact, kDI, xproj_W + wd * kXDbl * kDI, xdbl, kXDbl, 0,
                 nullptr, nullptr, 0, kRows, kXDbl, kDI, EP_NONE, 0, 1);
            // dt-proj + softplus: (788,48) x (1536,48)^T -> dtb
            gemm(xdbl, kXDbl, dt_W + wd * kDI * kDtRank, dtb, kDI, 0,
                 dt_b + wd * kDI, nullptr, 0, kRows, kDI, kDtRank,
                 EP_BIAS_SOFTPLUS, 0, 1);
            // selective scan
            scan_kernel<<<dim3(kDI / 256, kB), blk, 0, stream>>>(
                uact, dtb, xdbl, A_log + wd * kDI * kNState, ysc);
            // gating
            gate_kernel<<<(kRows * kDI + 255) / 256, blk, 0, stream>>>(
                ysc, uact, xz, Dp + wd * kDI);
            // out-proj into concat half; backward direction row-reversed
            gemm(ysc, kDI, out_W + wd * kDim * kDI, cat, 2 * kDim, dir * kDim,
                 nullptr, nullptr, 0, kRows, kDim, kDI, EP_NONE, dir, kL);
        }
        // merge + bias + residual (into h_)
        gemm(cat, 2 * kDim, merge_W + (long)d * kDim * 2 * kDim, h_, kDim, 0,
             merge_b + (long)d * kDim, h_, kDim,
             kRows, kDim, 2 * kDim, EP_BIAS_RESID, 0, 1);
    }

    // ---- final LN on cls rows only -> output (4,768) ----
    ln_kernel<<<kB, blk, 0, stream>>>(h_, (long)kL * kDim, fn_w, fn_b, out, kDim);
}
